// AttentionBlock_7773890806168
// MI455X (gfx1250) — matrix operand order repaired and checked by reading, not compiled, not run on hardware
//
#include <hip/hip_runtime.h>

typedef __attribute__((ext_vector_type(16))) _Float16 v16h;
typedef __attribute__((ext_vector_type(8)))  _Float16 v8h;
typedef __attribute__((ext_vector_type(8)))  float    v8f;
typedef __attribute__((ext_vector_type(4)))  int      v4i;

// Explicit LDS (address_space(3)) types so LDS accesses lower to ds_* ops,
// never flat_* (flat burns LOADcnt+DScnt and the aperture check).
typedef __attribute__((address_space(3))) _Float16 lds_h;
typedef __attribute__((address_space(3))) v8h      lds_v8h;

#ifndef __has_builtin
#define __has_builtin(x) 0
#endif
#if __has_builtin(__builtin_amdgcn_global_load_async_to_lds_b128) && \
    __has_builtin(__builtin_amdgcn_s_wait_asynccnt)
#define USE_ASYNC 1
#else
#define USE_ASYNC 0
#endif

#if USE_ASYNC
typedef __attribute__((address_space(1))) v4i* as1_v4i_p;  // global int4*
typedef __attribute__((address_space(3))) v4i* as3_v4i_p;  // LDS int4*
#endif

// ---------------------------------------------------------------------------
// CDNA5 data movers: async global->LDS copy (16B per lane), ASYNCcnt tracked.
// ---------------------------------------------------------------------------
__device__ __forceinline__ void copy16_to_lds(const _Float16* g, lds_h* l) {
#if USE_ASYNC
  __builtin_amdgcn_global_load_async_to_lds_b128(
      (as1_v4i_p)(_Float16*)g, (as3_v4i_p)l, 0, 0);
#else
  *(lds_v8h*)l = *(const v8h*)g;
#endif
}

template <int N>
__device__ __forceinline__ void wait_async_le() {
#if USE_ASYNC
  __builtin_amdgcn_s_wait_asynccnt(N);
#endif
}

// ---------------------------------------------------------------------------
// WMMA helpers (CDNA5 wave32, V_WMMA_F32_16X16X32_F16)
// ---------------------------------------------------------------------------
__device__ __forceinline__ v8f wmma_f16(v16h a, v16h b, v8f c) {
  return __builtin_amdgcn_wmma_f32_16x16x32_f16(
      false, a, false, b, (short)0, c, false, false);
}

// A-fragment (16M x 32K, f16), row-major [M, ld].
// ISA layout: lane m<16 holds row m, K {0..7}+{16..23}; lane m+16: K {8..15}+{24..31}.
__device__ __forceinline__ v16h load_frag_a(const _Float16* base, int ld) {
  const int lane = threadIdx.x & 31;
  const int r = lane & 15, hf = lane >> 4;
  const _Float16* p = base + r * ld + hf * 8;
  v8h lo = *(const v8h*)p;
  v8h hi = *(const v8h*)(p + 16);
  v16h f;
#pragma unroll
  for (int i = 0; i < 8; ++i) { f[i] = lo[i]; f[i + 8] = hi[i]; }
  return f;
}

__device__ __forceinline__ v16h load_frag_a_lds(const lds_h* base, int ld) {
  const int lane = threadIdx.x & 31;
  const int r = lane & 15, hf = lane >> 4;
  const lds_h* p = base + r * ld + hf * 8;
  v8h lo = *(const lds_v8h*)p;
  v8h hi = *(const lds_v8h*)(p + 16);
  v16h f;
#pragma unroll
  for (int i = 0; i < 8; ++i) { f[i] = lo[i]; f[i + 8] = hi[i]; }
  return f;
}

// B-fragment (32K x 16N, f16) from B^T stored row-major [N, ld].
// ISA layout: lane n<16 holds col n, K {0..7}+{16..23}; lane n+16: K {8..15}+{24..31}.
__device__ __forceinline__ v16h load_frag_b_lds(const lds_h* baseT, int ld) {
  const int lane = threadIdx.x & 31;
  const int r = lane & 15, hf = lane >> 4;
  const lds_h* p = baseT + r * ld + hf * 8;
  v8h lo = *(const lds_v8h*)p;
  v8h hi = *(const lds_v8h*)(p + 16);
  v16h f;
#pragma unroll
  for (int i = 0; i < 8; ++i) { f[i] = lo[i]; f[i + 8] = hi[i]; }
  return f;
}

// ---------------------------------------------------------------------------
// fp32 -> fp16 cast
// ---------------------------------------------------------------------------
__global__ void cast_f16_kernel(const float* __restrict__ src,
                                _Float16* __restrict__ dst, int n) {
  int i = blockIdx.x * blockDim.x + threadIdx.x;
  int stride = gridDim.x * blockDim.x;
  for (; i < n; i += stride) dst[i] = (_Float16)src[i];
}

// ---------------------------------------------------------------------------
// Shared 128x128 GEMM core, K=1024, double-buffered async LDS staging.
// LDS tiles padded to ld=40 halves (80B = 20 banks; gcd(20,64)=4 -> 16 distinct
// bank starts across fragment lanes, conflict-free 16B reads).
// ---------------------------------------------------------------------------
#define GLD 40  // padded LDS row stride (halves) for 32-wide K tiles

__device__ __forceinline__ void stage_ab(const _Float16* A, const _Float16* B,
                                         lds_h* as, lds_h* bs, int k0) {
  // 128 rows x 32 halves = 512 granules of 8 halves per matrix; 2 each/thread.
#pragma unroll
  for (int p = 0; p < 2; ++p) {
    const int g = threadIdx.x + p * 256;
    const int row = g >> 2, cc = (g & 3) * 8;
    copy16_to_lds(A + row * 1024 + k0 + cc, as + row * GLD + cc);
    copy16_to_lds(B + row * 1024 + k0 + cc, bs + row * GLD + cc);
  }
}

__device__ __forceinline__ void gemm_core(const _Float16* __restrict__ A,
                                          const _Float16* __restrict__ B,
                                          lds_h* smem, v8f c[4][2]) {
  const int wave = threadIdx.x >> 5;
  const int wm = (wave & 1) * 64, wn = (wave >> 1) * 32;
  lds_h* As[2] = { smem, smem + 128 * GLD };
  lds_h* Bs[2] = { smem + 2 * 128 * GLD, smem + 3 * 128 * GLD };

  stage_ab(A, B, As[0], Bs[0], 0);
  for (int kt = 0; kt < 32; ++kt) {
    const int cur = kt & 1;
    if (kt < 31) {
      stage_ab(A, B, As[cur ^ 1], Bs[cur ^ 1], (kt + 1) * 32);
      wait_async_le<4>();   // newest stage may stay in flight; current is done
    } else {
      wait_async_le<0>();
    }
    __syncthreads();        // everyone's portion of the current tile landed
    v16h a[4], b[2];
#pragma unroll
    for (int i = 0; i < 4; ++i)
      a[i] = load_frag_a_lds(As[cur] + (wm + 16 * i) * GLD, GLD);
#pragma unroll
    for (int j = 0; j < 2; ++j)
      b[j] = load_frag_b_lds(Bs[cur] + (wn + 16 * j) * GLD, GLD);
#pragma unroll
    for (int i = 0; i < 4; ++i)
#pragma unroll
      for (int j = 0; j < 2; ++j) c[i][j] = wmma_f16(a[i], b[j], c[i][j]);
    __syncthreads();        // reads done before this buffer is re-staged
  }
}

// ---------------------------------------------------------------------------
// QKV GEMM: C[8192,3072] = X @ W_in^T + b_in, scatter f16 -> qkv[3,B,H,S,64]
// ---------------------------------------------------------------------------
__global__ __launch_bounds__(256) void qkv_gemm_kernel(
    const _Float16* __restrict__ X, const _Float16* __restrict__ W,
    const float* __restrict__ bias, _Float16* __restrict__ qkv) {
  __shared__ __align__(16) _Float16 smem[4 * 128 * GLD];
  const int wave = threadIdx.x >> 5;
  const int m0 = blockIdx.y * 128 + (wave & 1) * 64;
  const int n0 = blockIdx.x * 128 + (wave >> 1) * 32;
  const int lane = threadIdx.x & 31;
  const int r = lane & 15, hf = lane >> 4;

  v8f c[4][2] = {};
  gemm_core(X + blockIdx.y * 128 * 1024, W + blockIdx.x * 128 * 1024,
            (lds_h*)smem, c);

  // C layout: lane r = column, VGPR v = row v + 8*hf.
#pragma unroll
  for (int j = 0; j < 2; ++j) {
    const int n = n0 + 16 * j + r;
    const float bn = bias[n];
    const int which = n >> 10, hh = (n >> 6) & 15, d = n & 63;
#pragma unroll
    for (int i = 0; i < 4; ++i)
#pragma unroll
      for (int v = 0; v < 8; ++v) {
        const int m = m0 + 16 * i + v + 8 * hf;
        const int bb = m >> 10, s = m & 1023;
        qkv[((((which * 8 + bb) * 16 + hh) << 10) + s) * 64 + d] =
            (_Float16)(c[i][j][v] + bn);
      }
  }
}

// ---------------------------------------------------------------------------
// Flash attention with post-softmax reweighting (softmax denominator cancels:
// out = (e^{s-max} * mult) @ V / sum(e^{s-max} * mult)).
// Block = 8 waves x 16 q-rows = 128 q rows of one (b,h); grid = 1024.
// K tile staged async into LDS (shared by all 8 waves); V staged transposed.
// LDS rows padded to 72 halves (36 banks) for conflict-free fragment reads.
// ---------------------------------------------------------------------------
#define FLD 72

__global__ __launch_bounds__(256) void flash_kernel(
    const _Float16* __restrict__ qkv, const float* __restrict__ mult,
    _Float16* __restrict__ attn) {
  __shared__ __align__(16) _Float16 lds_[64 * FLD /*K*/ + 64 * FLD /*V^T*/ +
                                         8 * 16 * FLD /*P per wave*/];
  lds_h* kt = (lds_h*)lds_;                 // [64 k][FLD] raw K rows, shared
  lds_h* vt = kt + 64 * FLD;                // [64 d][FLD] V^T, shared
  const int wave = threadIdx.x >> 5;
  lds_h* pt = kt + 2 * 64 * FLD + wave * 16 * FLD;  // [16 q][FLD], private
  const int lane = threadIdx.x & 31;
  const int r = lane & 15, hf = lane >> 4;

  const int qt = blockIdx.x & 7;
  const int h  = (blockIdx.x >> 3) & 15;
  const int b  = blockIdx.x >> 7;
  const int q0 = qt * 128 + wave * 16;

  const _Float16* qp = qkv + ((0 * 8 + b) * 16 + h) * (1024 * 64);
  const _Float16* kp = qkv + ((1 * 8 + b) * 16 + h) * (1024 * 64);
  const _Float16* vp = qkv + ((2 * 8 + b) * 16 + h) * (1024 * 64);
  const float* mp = mult + (b * 1024 + q0) * 1024;

  // Q fragments, pre-scaled by 1/sqrt(64) = 0.125 (exact in f16)
  v16h aq[2];
#pragma unroll
  for (int ks = 0; ks < 2; ++ks) {
    aq[ks] = load_frag_a(qp + q0 * 64 + ks * 32, 64);
#pragma unroll
    for (int e = 0; e < 16; ++e) aq[ks][e] = aq[ks][e] * (_Float16)0.125f;
  }

  v8f o[4] = {};
  float mrow[8], lrow[8];
#pragma unroll
  for (int i = 0; i < 8; ++i) { mrow[i] = -1e30f; lrow[i] = 0.f; }

  for (int k0 = 0; k0 < 1024; k0 += 64) {
    // Async-stage K tile: 64 rows x 64 halves = 512 granules, 2 per thread.
#pragma unroll
    for (int p = 0; p < 2; ++p) {
      const int g = threadIdx.x + p * 256;
      const int row = g >> 3, cc = (g & 7) * 8;
      copy16_to_lds(kp + (k0 + row) * 64 + cc, kt + row * FLD + cc);
    }
    // Meanwhile stage V^T (global -> VGPR -> transposed LDS scatter).
#pragma unroll
    for (int it = 0; it < 2; ++it) {
      const int idx = threadIdx.x + it * 256;
      const int kk = idx >> 3, dg = idx & 7;
      v8h vv = *(const v8h*)(vp + (k0 + kk) * 64 + dg * 8);
#pragma unroll
      for (int e = 0; e < 8; ++e) vt[(dg * 8 + e) * FLD + kk] = vv[e];
    }
    wait_async_le<0>();
    __syncthreads();

    // Scores: S[16 q][64 k] = Q @ K^T (K rows from LDS are B^T rows).
    v8f sc[4];
#pragma unroll
    for (int kc = 0; kc < 4; ++kc) {
      v8f z = {};
#pragma unroll
      for (int ks = 0; ks < 2; ++ks) {
        v16h bk = load_frag_b_lds(kt + (kc * 16) * FLD + ks * 32, FLD);
        z = wmma_f16(aq[ks], bk, z);
      }
      sc[kc] = z;
    }

    // Online max: row q = vgpr_i + 8*hf lives across the 16-lane half-group.
    float mx[8];
#pragma unroll
    for (int i = 0; i < 8; ++i)
      mx[i] = fmaxf(fmaxf(sc[0][i], sc[1][i]), fmaxf(sc[2][i], sc[3][i]));
#pragma unroll
    for (int off = 8; off; off >>= 1)
#pragma unroll
      for (int i = 0; i < 8; ++i) mx[i] = fmaxf(mx[i], __shfl_xor(mx[i], off, 16));

    float corr[8];
#pragma unroll
    for (int i = 0; i < 8; ++i) {
      float mn = fmaxf(mrow[i], mx[i]);
      corr[i] = __expf(mrow[i] - mn);
      mrow[i] = mn;
      lrow[i] *= corr[i];
    }
#pragma unroll
    for (int dc = 0; dc < 4; ++dc)
#pragma unroll
      for (int i = 0; i < 8; ++i) o[dc][i] *= corr[i];

    // P = e^{s-max} * mult ; accumulate row sums ; stash P (f16) for PV WMMA.
#pragma unroll
    for (int kc = 0; kc < 4; ++kc)
#pragma unroll
      for (int i = 0; i < 8; ++i) {
        const int q = i + 8 * hf;
        float p = __expf(sc[kc][i] - mrow[i]) * mp[q * 1024 + k0 + kc * 16 + r];
        pt[q * FLD + kc * 16 + r] = (_Float16)p;
        float rs = p;
#pragma unroll
        for (int off = 8; off; off >>= 1) rs += __shfl_xor(rs, off, 16);
        lrow[i] += rs;
      }

    // O += P @ V  (A = P from LDS, B = V via LDS-transposed V^T rows).
    v16h ap[2];
#pragma unroll
    for (int ks = 0; ks < 2; ++ks) ap[ks] = load_frag_a_lds(pt + ks * 32, FLD);
#pragma unroll
    for (int dc = 0; dc < 4; ++dc)
#pragma unroll
      for (int ks = 0; ks < 2; ++ks) {
        v16h bv = load_frag_b_lds(vt + (dc * 16) * FLD + ks * 32, FLD);
        o[dc] = wmma_f16(ap[ks], bv, o[dc]);
      }
    __syncthreads();   // tile reads done before next iteration re-stages
  }

  // out[q,d] = O / (l + eps), stored f16 as attn[b, s, h*64+d] for final GEMM.
#pragma unroll
  for (int dc = 0; dc < 4; ++dc)
#pragma unroll
    for (int i = 0; i < 8; ++i) {
      const int q = q0 + i + 8 * hf;
      const int d = dc * 16 + r;
      attn[(b * 1024 + q) * 1024 + h * 64 + d] =
          (_Float16)(o[dc][i] / (lrow[i] + 1e-15f));
    }
}

// ---------------------------------------------------------------------------
// Output projection: out[8192,1024] = A @ W_out^T + b_out (fp32 store)
// ---------------------------------------------------------------------------
__global__ __launch_bounds__(256) void out_gemm_kernel(
    const _Float16* __restrict__ A, const _Float16* __restrict__ W,
    const float* __restrict__ bias, float* __restrict__ out) {
  __shared__ __align__(16) _Float16 smem[4 * 128 * GLD];
  const int wave = threadIdx.x >> 5;
  const int m0 = blockIdx.y * 128 + (wave & 1) * 64;
  const int n0 = blockIdx.x * 128 + (wave >> 1) * 32;
  const int lane = threadIdx.x & 31;
  const int r = lane & 15, hf = lane >> 4;

  v8f c[4][2] = {};
  gemm_core(A + blockIdx.y * 128 * 1024, W + blockIdx.x * 128 * 1024,
            (lds_h*)smem, c);

#pragma unroll
  for (int j = 0; j < 2; ++j) {
    const int n = n0 + 16 * j + r;
    const float bn = bias[n];
#pragma unroll
    for (int i = 0; i < 4; ++i)
#pragma unroll
      for (int v = 0; v < 8; ++v) {
        const int m = m0 + 16 * i + v + 8 * hf;
        out[m * 1024 + n] = c[i][j][v] + bn;
      }
  }
}

// ---------------------------------------------------------------------------
// Launch: x, multipliers, W_in, b_in, W_out, b_out (all fp32).
// Workspace (88 MB): xh 16MB | W_in_h 6MB | W_out_h 2MB | qkv_h 48MB | attn_h 16MB
// ---------------------------------------------------------------------------
extern "C" void kernel_launch(void* const* d_in, const int* in_sizes, int n_in,
                              void* d_out, int out_size, void* d_ws, size_t ws_size,
                              hipStream_t stream) {
  (void)in_sizes; (void)n_in; (void)out_size; (void)ws_size;
  const float* x     = (const float*)d_in[0];
  const float* mult  = (const float*)d_in[1];
  const float* W_in  = (const float*)d_in[2];
  const float* b_in  = (const float*)d_in[3];
  const float* W_out = (const float*)d_in[4];
  const float* b_out = (const float*)d_in[5];
  float* out = (float*)d_out;

  char* ws = (char*)d_ws;
  _Float16* xh   = (_Float16*)(ws);
  _Float16* wih  = (_Float16*)(ws + (16u << 20));
  _Float16* woh  = (_Float16*)(ws + (22u << 20));
  _Float16* qkvh = (_Float16*)(ws + (24u << 20));
  _Float16* atth = (_Float16*)(ws + (72u << 20));

  cast_f16_kernel<<<2048, 256, 0, stream>>>(x, xh, 8 * 1024 * 1024);
  cast_f16_kernel<<<1024, 256, 0, stream>>>(W_in, wih, 3 * 1024 * 1024);
  cast_f16_kernel<<<512, 256, 0, stream>>>(W_out, woh, 1024 * 1024);

  qkv_gemm_kernel<<<dim3(24, 64), 256, 0, stream>>>(xh, wih, b_in, qkvh);
  flash_kernel<<<1024, 256, 0, stream>>>(qkvh, mult, atth);
  out_gemm_kernel<<<dim3(8, 64), 256, 0, stream>>>(atth, woh, b_out, out);
}
